// AttentionTCCNet_39178691674766
// MI455X (gfx1250) — compile-verified, matrix operand order checked by reading, not run on hardware
//
#include <hip/hip_runtime.h>
#include <hip/hip_fp16.h>
#include <cstdint>
#include <cstddef>

// ---------------------------------------------------------------------------
// Types for CDNA5 WMMA (wave32)
// ---------------------------------------------------------------------------
typedef __attribute__((ext_vector_type(16))) _Float16 v16h;
typedef __attribute__((ext_vector_type(8)))  _Float16 v8h;
typedef __attribute__((ext_vector_type(8)))  float    v8f;

#define TV_F 1e-5f
#define CT_F 1e-4f

__device__ __forceinline__ float sigm(float x) { return 1.0f / (1.0f + expf(-x)); }

// ---------------------------------------------------------------------------
// Implicit-GEMM 3x3 SAME conv on 7x7 images via v_wmma_f32_16x16x32_f16.
//   X : [Nimg][49][Cin]   f16 (NHWC)     W : [Cout][9][Cin] f16 (tap-major K)
//   Y : [Nimg][49][Cout]  OutT (NHWC)
// Block = 128 threads = 4 waves; wave w covers spatial tile n = 16w..16w+15.
// grid = (Nimg, Cout/16). FUSE: 1 = BN(scale,beta)+ReLU, 2 = +bias.
//
// Loop nest: tap (9) x c0 (Cin/64). Per-tap: pixel offset, validity, and B
// source pointer are computed once; invalid lanes read a zeroed pad buffer
// with stride 0, so the inner loop is branchless: 1+4 global_load_b128,
// 1+4 ds_store_b128, 8 ds_load_b128, 2 v_wmma per barrier pair.
// Fragments are two contiguous 16B runs per lane (ISA 7.12.2).
// ---------------------------------------------------------------------------
template <int FUSE, typename OutT>
__global__ __launch_bounds__(128)
void conv3x3_wmma(const _Float16* __restrict__ X,
                  const _Float16* __restrict__ W,
                  OutT* __restrict__ Y,
                  const float* __restrict__ scale,
                  const float* __restrict__ bias,
                  const _Float16* __restrict__ zpad,   // >=64 zero halves
                  int Cin, int Cout)
{
    const int img      = blockIdx.x;
    const int coutBase = blockIdx.y * 16;
    const int wave     = threadIdx.x >> 5;   // 0..3 spatial tile
    const int lane     = threadIdx.x & 31;
    const int K        = Cin * 9;
    const int nBase    = wave * 16;

    __shared__ _Float16 As[16][64];          // weight tile  (shared by 4 waves)
    __shared__ _Float16 Bs[4][16][64];       // im2col tile, n-major, per wave

    v8f acc = {};

    // B-build ownership: lane fills spatial row bn, k-section bsec (32 wide)
    const int bn   = lane & 15;
    const int bsec = lane >> 4;
    const int sp   = nBase + bn;
    const int oy   = sp / 7, ox = sp % 7;    // meaningful only if sp < 49

    // fragment gather constants
    const int hh = lane >> 4;
    const int mn = lane & 15;

    const _Float16* ximg = X + (size_t)img * 49 * Cin;

    // A-tile cooperative load assignment (8 threads x 16B per row)
    const int arow = threadIdx.x >> 3;
    const int acol = (threadIdx.x & 7) * 8;
    const _Float16* wsrc = W + (size_t)(coutBase + arow) * K + acol;

    v8h* const bdst = reinterpret_cast<v8h*>(&Bs[wave][bn][bsec * 32]);

    for (int tap = 0; tap < 9; ++tap) {
        const int dy = tap / 3 - 1;
        const int dx = tap % 3 - 1;
        const int py = oy + dy, px = ox + dx;
        const bool valid = (sp < 49) & ((unsigned)py < 7u) & ((unsigned)px < 7u);
        const v8h* xp = valid
            ? reinterpret_cast<const v8h*>(ximg + (size_t)(py * 7 + px) * Cin + bsec * 32)
            : reinterpret_cast<const v8h*>(zpad);
        const size_t xstep = valid ? 8 : 0;          // v8h units (64 halves)

        for (int c0 = 0; c0 < Cin; c0 += 64) {
            // ---- A tile slice (contiguous walk over [cout][9][Cin])
            *reinterpret_cast<v8h*>(&As[arow][acol]) =
                *reinterpret_cast<const v8h*>(wsrc);
            __builtin_prefetch(wsrc + 256, 0, 0);     // global_prefetch_b8
            wsrc += 64;
            // ---- B tile slice: 4x unconditional global_load_b128
            const v8h t0 = xp[0], t1 = xp[1], t2 = xp[2], t3 = xp[3];
            xp += xstep;
            bdst[0] = t0; bdst[1] = t1; bdst[2] = t2; bdst[3] = t3;

            __syncthreads();
#pragma unroll
            for (int half = 0; half < 2; ++half) {
                const int kb = half * 32;
                const v8h alo = *reinterpret_cast<const v8h*>(&As[mn][kb + 8 * hh]);
                const v8h ahi = *reinterpret_cast<const v8h*>(&As[mn][kb + 16 + 8 * hh]);
                const v8h blo = *reinterpret_cast<const v8h*>(&Bs[wave][mn][kb + 8 * hh]);
                const v8h bhi = *reinterpret_cast<const v8h*>(&Bs[wave][mn][kb + 16 + 8 * hh]);
                const v16h a = __builtin_shufflevector(alo, ahi, 0,1,2,3,4,5,6,7,
                                                       8,9,10,11,12,13,14,15);
                const v16h b = __builtin_shufflevector(blo, bhi, 0,1,2,3,4,5,6,7,
                                                       8,9,10,11,12,13,14,15);
                acc = __builtin_amdgcn_wmma_f32_16x16x32_f16(
                          false, a, false, b, (short)0, acc, false, false);
            }
            __syncthreads();
        }
    }

    // ---- epilogue: lane writes 8 consecutive channels at one pixel
    {
        const int n   = lane & 15;
        const int spo = nBase + n;
        if (spo < 49) {
            const int cb = coutBase + 8 * hh;
            float vals[8];
#pragma unroll
            for (int r = 0; r < 8; ++r) {
                float v = acc[r];
                if (FUSE == 1) {               // eval BN + ReLU
                    v = v * scale[cb + r] + bias[cb + r];
                    v = v > 0.0f ? v : 0.0f;
                } else if (FUSE == 2) {        // conv bias
                    v = v + bias[cb + r];
                }
                vals[r] = v;
            }
            OutT* yp = Y + ((size_t)img * 49 + spo) * Cout + cb;
            if constexpr (sizeof(OutT) == 2) {
                v8h o;
#pragma unroll
                for (int r = 0; r < 8; ++r) o[r] = (_Float16)vals[r];
                *reinterpret_cast<v8h*>(yp) = o;
            } else {
                v8f o;
#pragma unroll
                for (int r = 0; r < 8; ++r) o[r] = vals[r];
                *reinterpret_cast<v8f*>(yp) = o;
            }
        }
    }
}

// ---------------------------------------------------------------------------
// Weight convert+permute: w [Cout][Cin][9] f32 -> d [Cout][9][Cin] f16
// ---------------------------------------------------------------------------
__global__ void wperm_kernel(const float* __restrict__ w, _Float16* __restrict__ d,
                             int Cout, int Cin)
{
    const size_t N = (size_t)Cout * Cin * 9;
    for (size_t i = blockIdx.x * (size_t)blockDim.x + threadIdx.x; i < N;
         i += (size_t)gridDim.x * blockDim.x) {
        const int    cin  = (int)(i % Cin);
        const size_t r    = i / Cin;
        const int    tap  = (int)(r % 9);
        const size_t cout = r / 9;
        d[i] = (_Float16)w[(cout * Cin + cin) * 9 + tap];
    }
}

__global__ void bnprep_kernel(const float* __restrict__ p, float* __restrict__ scale,
                              float* __restrict__ beta, int C)
{
    int i = blockIdx.x * blockDim.x + threadIdx.x;
    if (i < C) {
        scale[i] = p[i] * rsqrtf(1.0f + 1e-5f);
        beta[i]  = p[C + i];
    }
}

__global__ void zero_kernel(float* __restrict__ p, int n)
{
    int i = blockIdx.x * blockDim.x + threadIdx.x;
    if (i < n) p[i] = 0.0f;
}

// input_x (B,C,T,7,7) f32 -> xr16 [(b*T+t)][hw][c] f16 (channel-minor)
__global__ void xr_kernel(const float* __restrict__ x, _Float16* __restrict__ xr)
{
    const size_t N = (size_t)128 * 2048 * 49;
    for (size_t i = blockIdx.x * (size_t)blockDim.x + threadIdx.x; i < N;
         i += (size_t)gridDim.x * blockDim.x) {
        const int    c  = (int)(i % 2048);
        const size_t r  = i / 2048;
        const int    hw = (int)(r % 49);
        const int    bt = (int)(r / 49);
        const int    b  = bt / 16, t = bt % 16;
        xr[i] = (_Float16)x[(((size_t)b * 2048 + c) * 16 + t) * 49 + hw];
    }
}

// mask conv: 512->1, 3x3 SAME, sigmoid; Y2 channel-minor [bt][hw][512]
__global__ void maskconv_kernel(const _Float16* __restrict__ Y2,
                                const float* __restrict__ w3,
                                float* __restrict__ mask)
{
    const int idx = blockIdx.x * 64 + threadIdx.x;
    if (idx >= 128 * 49) return;
    const int bt = idx / 49, hw = idx % 49;
    const int oy = hw / 7, ox = hw % 7;
    const _Float16* yb = Y2 + (size_t)bt * 49 * 512;
    float s = 0.0f;
    for (int r = 0; r < 3; ++r) {
        const int py = oy + r - 1;
        if (py < 0 || py >= 7) continue;
        for (int q = 0; q < 3; ++q) {
            const int px = ox + q - 1;
            if (px < 0 || px >= 7) continue;
            const _Float16* yp = yb + (size_t)(py * 7 + px) * 512;
            const int tap = r * 3 + q;
            for (int c = 0; c < 512; ++c)
                s += (float)yp[c] * w3[c * 9 + tap];
        }
    }
    mask[idx] = 1.0f / (1.0f + expf(-s));
}

// TV + contrast losses, single block, deterministic tree reduction
__global__ void losses_kernel(const float* __restrict__ mask,
                              float* __restrict__ out_tv, float* __restrict__ out_ct)
{
    __shared__ float stv[256], sct[256];
    float tv = 0.0f, ct = 0.0f;
    for (int i = threadIdx.x; i < 128 * 49; i += 256) {
        const int bt = i / 49, hw = i % 49;
        const int y = hw / 7, x = hw % 7;
        const float m = mask[bt * 49 + hw];
        if (x < 6) tv += fabsf(mask[bt * 49 + y * 7 + x + 1] - m);
        if (y < 6) tv += fabsf(mask[bt * 49 + (y + 1) * 7 + x] - m);
        const float mA = m > 0.5f ? 1.0f : 0.0f;
        const float mB = m < 0.5f ? 1.0f : 0.0f;
        ct += m * mB - m * mA;
    }
    stv[threadIdx.x] = tv; sct[threadIdx.x] = ct;
    __syncthreads();
    for (int s = 128; s > 0; s >>= 1) {
        if (threadIdx.x < s) { stv[threadIdx.x] += stv[threadIdx.x + s];
                               sct[threadIdx.x] += sct[threadIdx.x + s]; }
        __syncthreads();
    }
    if (threadIdx.x == 0) {
        *out_tv = TV_F * stv[0];
        *out_ct = (CT_F * 0.5f / 8.0f) * sct[0];
    }
}

// mx = mask * x; mx16 [bt][c][49] (channel-major), meanx [b][hw][2048] (NHWC)
__global__ void mx_mean_kernel(const float* __restrict__ x, const float* __restrict__ mask,
                               _Float16* __restrict__ mx, _Float16* __restrict__ meanx)
{
    const size_t N = (size_t)8 * 2048 * 49;
    for (size_t i = blockIdx.x * (size_t)blockDim.x + threadIdx.x; i < N;
         i += (size_t)gridDim.x * blockDim.x) {
        const int    hw = (int)(i % 49);
        const size_t r  = i / 49;
        const int    c  = (int)(r % 2048);
        const int    b  = (int)(r / 2048);
        float s = 0.0f;
        for (int t = 0; t < 16; ++t) {
            const float v = mask[((size_t)(b * 16 + t)) * 49 + hw] *
                            x[(((size_t)b * 2048 + c) * 16 + t) * 49 + hw];
            mx[(((size_t)(b * 16 + t)) * 2048 + c) * 49 + hw] = (_Float16)v;
            s += v;
        }
        meanx[((size_t)b * 49 + hw) * 2048 + c] = (_Float16)(s * (1.0f / 16.0f));
    }
}

// att_fea[bt] = (1/49) * sum_{c,hw} mx[bt,c,hw] * att_fw[c]
__global__ void attfea_kernel(const _Float16* __restrict__ mx,
                              const float* __restrict__ att_fw,
                              float* __restrict__ att_fea)
{
    __shared__ float sbuf[256];
    const int bt = blockIdx.x;
    const _Float16* base = mx + (size_t)bt * 2048 * 49;
    float s = 0.0f;
    for (int i = threadIdx.x; i < 2048 * 49; i += 256)
        s += (float)base[i] * att_fw[i / 49];
    sbuf[threadIdx.x] = s;
    __syncthreads();
    for (int q = 128; q > 0; q >>= 1) {
        if (threadIdx.x < q) sbuf[threadIdx.x] += sbuf[threadIdx.x + q];
        __syncthreads();
    }
    if (threadIdx.x == 0) att_fea[bt] = sbuf[0] * (1.0f / 49.0f);
}

// per-batch: att_h = mean_hw(h) . att_hw ; w = softmax(att_fea + att_h)
// h channel-minor [b][hw][512]
__global__ void softmax_kernel(const _Float16* __restrict__ h,
                               const float* __restrict__ att_hw,
                               const float* __restrict__ att_fea,
                               float* __restrict__ wbuf,
                               float* __restrict__ ws_out)
{
    __shared__ float sbuf[256];
    const int b = blockIdx.x;
    const _Float16* hb = h + (size_t)b * 49 * 512;
    float s = 0.0f;
    for (int i = threadIdx.x; i < 512 * 49; i += 256)
        s += (float)hb[i] * att_hw[i & 511];
    sbuf[threadIdx.x] = s;
    __syncthreads();
    for (int q = 128; q > 0; q >>= 1) {
        if (threadIdx.x < q) sbuf[threadIdx.x] += sbuf[threadIdx.x + q];
        __syncthreads();
    }
    if (threadIdx.x == 0) {
        const float atth = sbuf[0] * (1.0f / 49.0f);
        float l[16], mx = -3.4e38f;
        for (int t = 0; t < 16; ++t) { l[t] = att_fea[b * 16 + t] + atth; mx = fmaxf(mx, l[t]); }
        float sum = 0.0f;
        for (int t = 0; t < 16; ++t) { l[t] = expf(l[t] - mx); sum += l[t]; }
        for (int t = 0; t < 16; ++t) {
            const float w = l[t] / sum;
            wbuf[b * 16 + t] = w;
            if (ws_out) ws_out[b * 16 + t] = w;
        }
    }
}

// zin channel-minor [b][hw][2560]: ch<2048 -> weighted frame; else h (NHWC)
__global__ void zin_kernel(const _Float16* __restrict__ mx, const _Float16* __restrict__ h,
                           const float* __restrict__ wbuf, _Float16* __restrict__ zin)
{
    const size_t N = (size_t)8 * 49 * 2560;
    for (size_t i = blockIdx.x * (size_t)blockDim.x + threadIdx.x; i < N;
         i += (size_t)gridDim.x * blockDim.x) {
        const int    ch  = (int)(i % 2560);
        const size_t pix = i / 2560;
        const int    hw  = (int)(pix % 49);
        const int    b   = (int)(pix / 49);
        if (ch < 2048) {
            float s = 0.0f;
            for (int t = 0; t < 16; ++t)
                s += (float)mx[(((size_t)(b * 16 + t)) * 2048 + ch) * 49 + hw] * wbuf[b * 16 + t];
            zin[i] = (_Float16)s;
        } else {
            zin[i] = h[((size_t)b * 49 + hw) * 512 + (ch - 2048)];
        }
    }
}

// LSTM gates; z [b][hw][2048] NHWC, cbuf/h16 [b][hw][512] NHWC
__global__ __launch_bounds__(64)
void gates_kernel(const float* __restrict__ z, float* __restrict__ cbuf,
                  _Float16* __restrict__ h16, float* __restrict__ outs_sum)
{
    __shared__ float sred[64];
    const int b  = blockIdx.x / 512;
    const int ch = blockIdx.x % 512;
    const int hw = threadIdx.x;
    float h2v = 0.0f;
    if (hw < 49) {
        const size_t base = ((size_t)b * 49 + hw) * 2048;
        const float iv = z[base + ch       ];
        const float fv = z[base + ch +  512];
        const float ov = z[base + ch + 1024];
        const float gv = z[base + ch + 1536];
        const size_t ci = ((size_t)b * 49 + hw) * 512 + ch;
        const float c2 = sigm(fv) * cbuf[ci] + sigm(iv) * tanhf(gv);
        const float h2 = sigm(ov) * tanhf(c2);
        cbuf[ci] = c2;
        h16[ci]  = (_Float16)h2;
        h2v = h2;
    }
    sred[threadIdx.x] = h2v;
    __syncthreads();
    for (int s = 32; s > 0; s >>= 1) {
        if (threadIdx.x < s) sred[threadIdx.x] += sred[threadIdx.x + s];
        __syncthreads();
    }
    if (threadIdx.x == 0)
        outs_sum[b * 512 + ch] += sred[0] * (1.0f / (49.0f * 16.0f));
}

// final_output[b][k] = outs_sum[b] . fc_w[k] + fc_b[k]
__global__ void fc_kernel(const float* __restrict__ outs_sum, const float* __restrict__ fc_w,
                          const float* __restrict__ fc_b, float* __restrict__ out)
{
    const int idx = threadIdx.x;
    if (idx < 24) {
        const int b = idx / 3, k = idx % 3;
        float s = fc_b[k];
        for (int c = 0; c < 512; ++c)
            s += outs_sum[b * 512 + c] * fc_w[k * 512 + c];
        out[b * 3 + k] = s;
    }
}

// ---------------------------------------------------------------------------
// Host launcher
// ---------------------------------------------------------------------------
static inline char* bump(char*& p, size_t bytes)
{
    char* r = (char*)(((uintptr_t)p + 255) & ~(uintptr_t)255);
    p = r + bytes;
    return r;
}

extern "C" void kernel_launch(void* const* d_in, const int* in_sizes, int n_in,
                              void* d_out, int out_size, void* d_ws, size_t ws_size,
                              hipStream_t stream)
{
    (void)in_sizes; (void)n_in; (void)out_size; (void)ws_size;

    const float* input_x = (const float*)d_in[0];
    const float* mask_w1 = (const float*)d_in[1];
    const float* mask_bn1= (const float*)d_in[2];
    const float* mask_w2 = (const float*)d_in[3];
    const float* mask_bn2= (const float*)d_in[4];
    const float* mask_w3 = (const float*)d_in[5];
    const float* h0_w1   = (const float*)d_in[6];
    const float* h0_bn1  = (const float*)d_in[7];
    const float* h0_w2   = (const float*)d_in[8];
    const float* h0_bn2  = (const float*)d_in[9];
    const float* c0_w1   = (const float*)d_in[10];
    const float* c0_bn1  = (const float*)d_in[11];
    const float* c0_w2   = (const float*)d_in[12];
    const float* c0_bn2  = (const float*)d_in[13];
    const float* att_fw  = (const float*)d_in[14];
    const float* att_hw  = (const float*)d_in[15];
    const float* lstm_w  = (const float*)d_in[16];
    const float* lstm_b  = (const float*)d_in[17];
    const float* fc_w    = (const float*)d_in[18];
    const float* fc_b    = (const float*)d_in[19];

    float* outf      = (float*)d_out;
    float* out_final = outf;           // 24
    float* out_ws    = outf + 24;      // 128
    float* out_mask  = outf + 152;     // 6272
    float* out_tv    = outf + 6424;    // 1
    float* out_ct    = outf + 6425;    // 1

    // ---- workspace carve-up
    char* p = (char*)d_ws;
    _Float16* xr16   = (_Float16*)bump(p, (size_t)128*2048*49 * 2);
    _Float16* w1m    = (_Float16*)bump(p, (size_t)1024*2048*9 * 2);
    _Float16* w2m    = (_Float16*)bump(p, (size_t)512*1024*9  * 2);
    _Float16* wh1    = (_Float16*)bump(p, (size_t)1024*2048*9 * 2);
    _Float16* wh2    = (_Float16*)bump(p, (size_t)512*1024*9  * 2);
    _Float16* wc1    = (_Float16*)bump(p, (size_t)1024*2048*9 * 2);
    _Float16* wc2    = (_Float16*)bump(p, (size_t)512*1024*9  * 2);
    _Float16* wl16   = (_Float16*)bump(p, (size_t)2048*2560*9 * 2);
    _Float16* y1     = (_Float16*)bump(p, (size_t)128*1024*49 * 2);
    _Float16* y2     = (_Float16*)bump(p, (size_t)128*512*49  * 2);
    _Float16* mx16   = (_Float16*)bump(p, (size_t)128*2048*49 * 2);
    _Float16* meanx  = (_Float16*)bump(p, (size_t)8*2048*49   * 2);
    _Float16* t1024  = (_Float16*)bump(p, (size_t)8*1024*49   * 2);
    _Float16* hbuf   = (_Float16*)bump(p, (size_t)8*512*49    * 2);
    _Float16* zin16  = (_Float16*)bump(p, (size_t)8*2560*49   * 2);
    _Float16* zpad   = (_Float16*)bump(p, 64 * 2);         // 128B of zeros
    float*    cbuf   = (float*)bump(p, (size_t)8*512*49  * 4);
    float*    zbuf   = (float*)bump(p, (size_t)8*2048*49 * 4);
    float*    attfea = (float*)bump(p, 128 * 4);
    float*    wbuf   = (float*)bump(p, 128 * 4);
    float*    osum   = (float*)bump(p, 4096 * 4);
    float*    s_m1   = (float*)bump(p, 1024 * 4);
    float*    b_m1   = (float*)bump(p, 1024 * 4);
    float*    s_m2   = (float*)bump(p, 512 * 4);
    float*    b_m2   = (float*)bump(p, 512 * 4);
    float*    s_h1   = (float*)bump(p, 1024 * 4);
    float*    b_h1   = (float*)bump(p, 1024 * 4);
    float*    s_h2   = (float*)bump(p, 512 * 4);
    float*    b_h2   = (float*)bump(p, 512 * 4);
    float*    s_c1   = (float*)bump(p, 1024 * 4);
    float*    b_c1   = (float*)bump(p, 1024 * 4);
    float*    s_c2   = (float*)bump(p, 512 * 4);
    float*    b_c2   = (float*)bump(p, 512 * 4);

    // ---- zero pad buffer (used by all convs for out-of-bounds taps)
    zero_kernel<<<1, 32, 0, stream>>>((float*)zpad, 32);

    // ---- weight convert + tap-major permute ([Cout][Cin][9] -> [Cout][9][Cin])
    auto permw = [&](const float* s, _Float16* d, int Cout, int Cin) {
        size_t n = (size_t)Cout * Cin * 9;
        int blocks = (int)((n + 255) / 256);
        if (blocks > 65535) blocks = 65535;
        wperm_kernel<<<blocks, 256, 0, stream>>>(s, d, Cout, Cin);
    };
    permw(mask_w1, w1m, 1024, 2048);
    permw(mask_w2, w2m, 512, 1024);
    permw(h0_w1,   wh1, 1024, 2048);
    permw(h0_w2,   wh2, 512, 1024);
    permw(c0_w1,   wc1, 1024, 2048);
    permw(c0_w2,   wc2, 512, 1024);
    permw(lstm_w,  wl16, 2048, 2560);
    bnprep_kernel<<<4, 256, 0, stream>>>(mask_bn1, s_m1, b_m1, 1024);
    bnprep_kernel<<<2, 256, 0, stream>>>(mask_bn2, s_m2, b_m2, 512);
    bnprep_kernel<<<4, 256, 0, stream>>>(h0_bn1,   s_h1, b_h1, 1024);
    bnprep_kernel<<<2, 256, 0, stream>>>(h0_bn2,   s_h2, b_h2, 512);
    bnprep_kernel<<<4, 256, 0, stream>>>(c0_bn1,   s_c1, b_c1, 1024);
    bnprep_kernel<<<2, 256, 0, stream>>>(c0_bn2,   s_c2, b_c2, 512);

    // ---- input permute + cast (NCHW -> NHWC f16)
    xr_kernel<<<8192, 256, 0, stream>>>(input_x, xr16);

    // ---- mask branch: 2048 -> 1024 -> 512 -> 1(sigmoid)
    conv3x3_wmma<1, _Float16><<<dim3(128, 64), 128, 0, stream>>>(
        xr16, w1m, y1, s_m1, b_m1, zpad, 2048, 1024);
    conv3x3_wmma<1, _Float16><<<dim3(128, 32), 128, 0, stream>>>(
        y1, w2m, y2, s_m2, b_m2, zpad, 1024, 512);
    maskconv_kernel<<<98, 64, 0, stream>>>(y2, mask_w3, out_mask);
    losses_kernel<<<1, 256, 0, stream>>>(out_mask, out_tv, out_ct);

    // ---- mx = mask * x, mean over T; attention feature logits
    mx_mean_kernel<<<3136, 256, 0, stream>>>(input_x, out_mask, mx16, meanx);
    attfea_kernel<<<128, 256, 0, stream>>>(mx16, att_fw, attfea);

    // ---- h0 / c0 initial states (two_conv each)
    conv3x3_wmma<1, _Float16><<<dim3(8, 64), 128, 0, stream>>>(
        meanx, wh1, t1024, s_h1, b_h1, zpad, 2048, 1024);
    conv3x3_wmma<1, _Float16><<<dim3(8, 32), 128, 0, stream>>>(
        t1024, wh2, hbuf, s_h2, b_h2, zpad, 1024, 512);
    conv3x3_wmma<1, _Float16><<<dim3(8, 64), 128, 0, stream>>>(
        meanx, wc1, t1024, s_c1, b_c1, zpad, 2048, 1024);
    conv3x3_wmma<1, float><<<dim3(8, 32), 128, 0, stream>>>(
        t1024, wc2, cbuf, s_c2, b_c2, zpad, 1024, 512);

    // ---- LSTM scan
    zero_kernel<<<16, 256, 0, stream>>>(osum, 4096);
    for (int step = 0; step < 16; ++step) {
        softmax_kernel<<<8, 256, 0, stream>>>(hbuf, att_hw, attfea, wbuf,
                                              step == 15 ? out_ws : nullptr);
        zin_kernel<<<3920, 256, 0, stream>>>(mx16, hbuf, wbuf, zin16);
        conv3x3_wmma<2, float><<<dim3(8, 128), 128, 0, stream>>>(
            zin16, wl16, zbuf, nullptr, lstm_b, zpad, 2560, 2048);
        gates_kernel<<<4096, 64, 0, stream>>>(zbuf, cbuf, hbuf, osum);
    }

    // ---- final FC
    fc_kernel<<<1, 32, 0, stream>>>(osum, fc_w, fc_b, out_final);
}
